// GraphEncoding_12541304504494
// MI455X (gfx1250) — matrix operand order repaired and checked by reading, not compiled, not run on hardware
//
#include <hip/hip_runtime.h>
#include <hip/hip_bf16.h>

// CDNA5 / gfx1250, wave32. All matrix contractions via v_wmma_f32_16x16x32_f16.
// Weights + GEMM activations pre-converted to f16 so the WMMA hot loops issue
// pure 16B/32B loads with no f32->f16 cvt VALU traffic.

typedef __attribute__((ext_vector_type(16))) _Float16 v16h;
typedef __attribute__((ext_vector_type(8)))  _Float16 v8h;
typedef __attribute__((ext_vector_type(8)))  float    v8f;

#define N_NODES 4096
#define HDIM    128
#define NHEADS  2
#define COUT    64
#define NEG_SLOPE 0.2f

__device__ __forceinline__ float lrelu(float v) { return v > 0.f ? v : NEG_SLOPE * v; }

// Order-preserving float->uint map for atomicMax on floats (any sign).
__device__ __forceinline__ unsigned mapf(float f) {
    unsigned u = __float_as_uint(f);
    return (u & 0x80000000u) ? ~u : (u | 0x80000000u);
}
__device__ __forceinline__ float unmapf(unsigned u) {
    return (u & 0x80000000u) ? __uint_as_float(u & 0x7FFFFFFFu) : __uint_as_float(~u);
}

// ---------------------------------------------------------------------------
// f32 -> f16 bulk convert (count must be a multiple of 8; coalesced).
// ---------------------------------------------------------------------------
__global__ __launch_bounds__(256) void cvt_f16_kernel(
    const float* __restrict__ src, _Float16* __restrict__ dst, int count)
{
    const int g = (blockIdx.x * blockDim.x + threadIdx.x) * 8;
    if (g >= count) return;
    const float4* s4 = (const float4*)(src + g);
    float4 a = s4[0], b = s4[1];
    v8h h;
    h[0] = (_Float16)a.x; h[1] = (_Float16)a.y;
    h[2] = (_Float16)a.z; h[3] = (_Float16)a.w;
    h[4] = (_Float16)b.x; h[5] = (_Float16)b.y;
    h[6] = (_Float16)b.z; h[7] = (_Float16)b.w;
    *(v8h*)(dst + g) = h;
}

// ---------------------------------------------------------------------------
// GEMM: C[N,128] = A[N,128] (f16 row-major) x W^T, Wh [128,128] f16 row-major
// (out_features x in_features): C[i][o] = sum_k A[i][k]*Wh[o][k].
// One wave computes a 16-row x 128-col strip; K loop of 4x (16x16x32 WMMA).
// A-frag: two contiguous v8h loads (K {akb..+7} U {akb+8..+15}).
// B-frag: single contiguous v16h load (K = hi*16..+15 contiguous in row).
// Writes: f32 row-major (+bias) and/or f16 TRANSPOSED Ct[128][N] (attention V;
// the f32 D layout gives each lane 8 consecutive rows of one column => one
// 16B packed-f16 store per tile).
// ---------------------------------------------------------------------------
__global__ __launch_bounds__(128) void gemm128_kernel(
    const _Float16* __restrict__ A, const _Float16* __restrict__ Wh,
    const float* __restrict__ bias, int useBias,
    float* __restrict__ Crm, int writeRm,
    _Float16* __restrict__ Ct, int writeTr)
{
    const int lane = threadIdx.x & 31;
    const int wid  = blockIdx.x * (blockDim.x >> 5) + (threadIdx.x >> 5);
    const int i0   = wid * 16;
    const int m    = lane & 15;       // A-frag row / B-frag column / D column
    const int hi   = lane >> 4;       // lane half selects K sub-ranges
    const int akb  = hi * 16;         // A: K in {akb..akb+15}

    const _Float16* __restrict__ arow = A + (i0 + m) * HDIM;

    v8f acc[8] = {};

    #pragma unroll
    for (int kc = 0; kc < HDIM; kc += 32) {
        v8h alo = *(const v8h*)(arow + kc + akb);
        v8h ahi = *(const v8h*)(arow + kc + akb + 8);
        v16h af = __builtin_shufflevector(alo, ahi,
                    0, 1, 2, 3, 4, 5, 6, 7, 8, 9, 10, 11, 12, 13, 14, 15);

        #pragma unroll
        for (int t = 0; t < 8; t++) {
            const v16h bf = *(const v16h*)(Wh + (t * 16 + m) * HDIM + kc + hi * 16);
            acc[t] = __builtin_amdgcn_wmma_f32_16x16x32_f16(
                false, af, false, bf, (short)0, acc[t], false, false);
        }
    }

    // ---- epilogue: D layout lane holds rows mbase..mbase+7 of column m ----
    const int mbase = hi * 8;
    #pragma unroll
    for (int t = 0; t < 8; t++) {
        const int col = t * 16 + m;
        const float bv = useBias ? bias[col] : 0.f;
        if (writeRm) {
            #pragma unroll
            for (int v = 0; v < 8; v++)
                Crm[(i0 + mbase + v) * HDIM + col] = acc[t][v] + bv;
        }
        if (writeTr) {
            v8h hv;
            #pragma unroll
            for (int v = 0; v < 8; v++) hv[v] = (_Float16)(acc[t][v] + bv);
            *(v8h*)(Ct + col * N_NODES + i0 + mbase) = hv;  // 16B contiguous
        }
    }
}

// ---------------------------------------------------------------------------
// Per-(head,node) attention scalars: s = <xh_head, a_src>, d = <xh_head, a_dst>
// plus per-head max_j s_j via order-preserving uint atomicMax.
// ---------------------------------------------------------------------------
__global__ __launch_bounds__(256) void attprep_kernel(
    const float* __restrict__ xh, const float* __restrict__ asrc,
    const float* __restrict__ adst, float* __restrict__ sv,
    float* __restrict__ dv, unsigned* __restrict__ smax_u)
{
    const int tid = threadIdx.x;
    const int g   = blockIdx.x * blockDim.x + tid;
    const int h   = g & 1;
    const int n   = g >> 1;

    const float4* r4 = (const float4*)(xh + n * HDIM + h * COUT);
    const float4* a4 = (const float4*)(asrc + h * COUT);
    const float4* d4 = (const float4*)(adst + h * COUT);
    float s = 0.f, d = 0.f;
    #pragma unroll
    for (int c = 0; c < COUT / 4; c++) {
        float4 xv = r4[c], av = a4[c], dvv = d4[c];
        s += xv.x * av.x + xv.y * av.y + xv.z * av.z + xv.w * av.w;
        d += xv.x * dvv.x + xv.y * dvv.y + xv.z * dvv.z + xv.w * dvv.w;
    }
    sv[h * N_NODES + n] = s;
    dv[h * N_NODES + n] = d;

    __shared__ float red[256];
    red[tid] = s;
    __syncthreads();
    // parity-preserving max reduce (even offsets keep heads separated)
    for (int off = 128; off >= 2; off >>= 1) {
        if (tid < off) red[tid] = fmaxf(red[tid], red[tid + off]);
        __syncthreads();
    }
    if (tid < 2) atomicMax(&smax_u[tid], mapf(red[tid]));
}

__global__ void init_smax_kernel(unsigned* smax_u)
{
    if (threadIdx.x < 2) smax_u[threadIdx.x] = 0u;
}

// ---------------------------------------------------------------------------
// Attention apply + fused epilogue. One wave = (head, 16-row tile).
// P tile (16x32) built directly in WMMA A-fragment layout with f32 exp math;
// V tiles read as contiguous 32B/lane B-fragments from transposed f16 xhT.
// rowmax_i = lrelu(d_i + max_j s_j) is the exact row max (lrelu monotone).
// Epilogue: xout = r*z + (1-r)*relu(attnOut/rowsum + g_b) + xin.
// ---------------------------------------------------------------------------
__global__ __launch_bounds__(128) void attn_kernel(
    const float* __restrict__ sv, const float* __restrict__ dv,
    const unsigned* __restrict__ smax_u, const _Float16* __restrict__ vt,
    const float* __restrict__ zbuf, const float* __restrict__ xin,
    const float* __restrict__ gbias, const float* __restrict__ rptr,
    float* __restrict__ xout)
{
    const int lane  = threadIdx.x & 31;
    const int wid   = blockIdx.x * (blockDim.x >> 5) + (threadIdx.x >> 5);
    const int head  = wid >> 8;              // 256 row-tiles per head
    const int i0    = (wid & 255) * 16;
    const int m     = lane & 15;
    const int hi    = lane >> 4;
    const int kb    = hi * 16;               // A-frag K base
    const int kbase = hi * 16;               // B-frag K base

    const float d      = dv[head * N_NODES + i0 + m];
    const float smax   = unmapf(smax_u[head]);
    const float rowmax = lrelu(d + smax);
    const float* __restrict__ sp = sv + head * N_NODES;
    const _Float16* __restrict__ vbase = vt + head * COUT * N_NODES;

    v8f acc[4] = {};
    float psum = 0.f;

    for (int j = 0; j < N_NODES; j += 32) {
        const float4* s0p = (const float4*)(sp + j + kb);
        const float4* s1p = (const float4*)(sp + j + kb + 8);
        float4 s0 = s0p[0], s1 = s0p[1], s2 = s1p[0], s3 = s1p[1];
        float st[16] = { s0.x, s0.y, s0.z, s0.w, s1.x, s1.y, s1.z, s1.w,
                         s2.x, s2.y, s2.z, s2.w, s3.x, s3.y, s3.z, s3.w };
        v16h pf;
        #pragma unroll
        for (int e = 0; e < 16; e++) {
            float p = __expf(lrelu(d + st[e]) - rowmax);   // in (0,1]
            _Float16 ph = (_Float16)p;
            pf[e] = ph;
            psum += (float)ph;   // denominator from f16-rounded p (consistent)
        }
        #pragma unroll
        for (int t = 0; t < 4; t++) {
            const v16h* bp = (const v16h*)(vbase + (t * 16 + m) * N_NODES + j + kbase);
            acc[t] = __builtin_amdgcn_wmma_f32_16x16x32_f16(
                false, pf, false, *bp, (short)0, acc[t], false, false);
        }
    }

    // rowsum: lane L holds partial of row L%16; complement lives at L^16
    psum += __shfl_xor(psum, 16, 32);
    const int mbase = hi * 8;
    float rs[8];
    #pragma unroll
    for (int v = 0; v < 8; v++) rs[v] = __shfl(psum, mbase + v, 32);

    const float r = rptr[0];
    #pragma unroll
    for (int t = 0; t < 4; t++) {
        const int col = head * COUT + t * 16 + m;
        const float gb = gbias[col];
        #pragma unroll
        for (int v = 0; v < 8; v++) {
            const int row = i0 + mbase + v;
            const int idx = row * HDIM + col;
            float gat = acc[t][v] / rs[v] + gb;
            float gr  = gat > 0.f ? gat : 0.f;
            xout[idx] = r * zbuf[idx] + (1.f - r) * gr + xin[idx];
        }
    }
}

// ---------------------------------------------------------------------------
extern "C" void kernel_launch(void* const* d_in, const int* in_sizes, int n_in,
                              void* d_out, int out_size, void* d_ws, size_t ws_size,
                              hipStream_t stream)
{
    (void)in_sizes; (void)n_in; (void)out_size; (void)ws_size;

    const float* x    = (const float*)d_in[0];
    // d_in[1] = city_size (unused)
    const float* r1   = (const float*)d_in[2];
    const float* r2   = (const float*)d_in[3];
    const float* W1w  = (const float*)d_in[4];
    const float* W1b  = (const float*)d_in[5];
    const float* W2w  = (const float*)d_in[6];
    const float* W2b  = (const float*)d_in[7];
    const float* g1l  = (const float*)d_in[8];
    const float* g1as = (const float*)d_in[9];
    const float* g1ad = (const float*)d_in[10];
    const float* g1b  = (const float*)d_in[11];
    const float* g2l  = (const float*)d_in[12];
    const float* g2as = (const float*)d_in[13];
    const float* g2ad = (const float*)d_in[14];
    const float* g2b  = (const float*)d_in[15];
    float* out = (float*)d_out;

    char* ws = (char*)d_ws;
    size_t off = 0;
    auto alloc = [&](size_t bytes) -> char* {
        char* p = ws + off;
        off += (bytes + 255) & ~(size_t)255;
        return p;
    };
    float*     zbuf  = (float*)alloc((size_t)N_NODES * HDIM * 4);
    float*     xhrm  = (float*)alloc((size_t)N_NODES * HDIM * 4);
    _Float16*  xht   = (_Float16*)alloc((size_t)N_NODES * HDIM * 2);
    float*     sv    = (float*)alloc((size_t)NHEADS * N_NODES * 4);
    float*     dvv   = (float*)alloc((size_t)NHEADS * N_NODES * 4);
    unsigned*  smax  = (unsigned*)alloc(256);
    float*     x1    = (float*)alloc((size_t)N_NODES * HDIM * 4);
    _Float16*  x16   = (_Float16*)alloc((size_t)N_NODES * HDIM * 2); // layer in, f16
    _Float16*  w1h   = (_Float16*)alloc((size_t)HDIM * HDIM * 2);
    _Float16*  w2h   = (_Float16*)alloc((size_t)HDIM * HDIM * 2);
    _Float16*  g1lh  = (_Float16*)alloc((size_t)HDIM * HDIM * 2);
    _Float16*  g2lh  = (_Float16*)alloc((size_t)HDIM * HDIM * 2);

    const int NW = HDIM * HDIM;          // 16384
    const int NX = N_NODES * HDIM;       // 524288

    // One-time (per call) f16 conversions of weights and layer-1 input.
    cvt_f16_kernel<<<NW / (256 * 8), 256, 0, stream>>>(W1w, w1h, NW);
    cvt_f16_kernel<<<NW / (256 * 8), 256, 0, stream>>>(W2w, w2h, NW);
    cvt_f16_kernel<<<NW / (256 * 8), 256, 0, stream>>>(g1l, g1lh, NW);
    cvt_f16_kernel<<<NW / (256 * 8), 256, 0, stream>>>(g2l, g2lh, NW);
    cvt_f16_kernel<<<NX / (256 * 8), 256, 0, stream>>>(x, x16, NX);

    auto layer = [&](const _Float16* xin16, const float* xin,
                     const _Float16* Wh, const float* Wb,
                     const _Float16* glh, const float* gas, const float* gad,
                     const float* gb, const float* r, float* xo) {
        // z = xin @ W^T + b  (f32 row-major)
        gemm128_kernel<<<64, 128, 0, stream>>>(xin16, Wh, Wb, 1,
                                               zbuf, 1, (_Float16*)nullptr, 0);
        // xh = xin @ lin^T   (f32 row-major for dots, f16 transposed for V)
        gemm128_kernel<<<64, 128, 0, stream>>>(xin16, glh, (const float*)nullptr, 0,
                                               xhrm, 1, xht, 1);
        init_smax_kernel<<<1, 32, 0, stream>>>(smax);
        attprep_kernel<<<32, 256, 0, stream>>>(xhrm, gas, gad, sv, dvv, smax);
        attn_kernel<<<128, 128, 0, stream>>>(sv, dvv, smax, xht, zbuf, xin,
                                             gb, r, xo);
    };

    layer(x16, x, w1h, W1b, g1lh, g1as, g1ad, g1b, r1, x1);
    // layer-2 GEMM input in f16
    cvt_f16_kernel<<<NX / (256 * 8), 256, 0, stream>>>(x1, x16, NX);
    layer(x16, x1, w2h, W2b, g2lh, g2as, g2ad, g2b, r2, out);
}
